// Model_25855703122644
// MI455X (gfx1250) — matrix operand order repaired and checked by reading, not compiled, not run on hardware
//
#include <hip/hip_runtime.h>
#include <hip/hip_bf16.h>
#include <math.h>
#include <stdint.h>

#define NDRUG 20000
#define NPROT 20000
#define NEDGE 160000
#define F_IN  512
#define F_HID 256
#define F_OUT 128
#define SEMD  128
#define NSLOPE 0.2f

typedef unsigned short u16;
typedef __attribute__((ext_vector_type(16))) __bf16 v16bf;
typedef __attribute__((ext_vector_type(8)))  __bf16 v8bf;
typedef __attribute__((ext_vector_type(8)))  float  v8f;

static inline int ceil_div(int a, int b) { return (a + b - 1) / b; }

__device__ __forceinline__ u16 f2bf(float f) {
  union { float f; unsigned u; } c; c.f = f;
  return (u16)((c.u + 0x7FFFu + ((c.u >> 16) & 1u)) >> 16);
}

__device__ __forceinline__ float wave_sum(float v) {
  for (int off = 16; off > 0; off >>= 1) v += __shfl_down(v, off, 32);
  return v;
}

// ---------------- conversion / fill ----------------
__global__ void k_f32_to_bf16(const float* __restrict__ in, u16* __restrict__ out, int n) {
  int i = blockIdx.x * blockDim.x + threadIdx.x;
  if (i < n) out[i] = f2bf(in[i]);
}

// in: K x N row-major f32 -> out: N x K row-major bf16 (transpose + quantize)
__global__ void k_transpose_bf16(const float* __restrict__ in, u16* __restrict__ out, int K, int N) {
  int i = blockIdx.x * blockDim.x + threadIdx.x;
  if (i >= K * N) return;
  int k = i / N, n = i - k * N;
  out[(size_t)n * K + k] = f2bf(in[i]);
}

__global__ void k_fill(float* __restrict__ p, float v, int n) {
  int i = blockIdx.x * blockDim.x + threadIdx.x;
  if (i < n) p[i] = v;
}

// ---------------- WMMA GEMM: C[M,N] = A[M,K] (bf16) * Bt[N,K]^T (bf16) ----------------
// One wave computes a 32x32 C tile: 2 A fragments x 2 B fragments -> 4 f32 accumulators.
// Per 32-k step: 4x v_wmma_f32_16x16x32_bf16 against 4KB of fragment loads
// (~16 FLOP/byte, 2x the intensity of a 16x16 tile).
// A fragment (16x32): lane r=lane&15 -> row; lanes<16: K=k0+0..7 & k0+16..23,
// lanes>=16: K=k0+8..15 & k0+24..31 (two contiguous 16B loads).
// B fragment (32x16): lane -> column; same K split as A (two 16B loads)
// from the pre-transposed Bt row.
__global__ void k_gemm_bf16(const u16* __restrict__ A, const u16* __restrict__ Bt,
                            float* __restrict__ C, int M, int N, int K) {
  int wave = (blockIdx.x * blockDim.x + threadIdx.x) >> 5;
  int lane = threadIdx.x & 31;
  int ntiles = N >> 5;                 // 32-wide column tiles
  int tm = (wave / ntiles) << 5;
  int tn = (wave % ntiles) << 5;
  if (tm >= M) return;                 // uniform per wave: EXEC stays all-ones for WMMA
  int r  = lane & 15;
  int hi = lane >> 4;
  int ka = hi ? 8 : 0;
  int kb = hi ? 16 : 0;
  const u16* arow0 = A  + (size_t)(tm + r) * K;
  const u16* arow1 = A  + (size_t)(tm + 16 + r) * K;
  const u16* brow0 = Bt + (size_t)(tn + r) * K;
  const u16* brow1 = Bt + (size_t)(tn + 16 + r) * K;
  v8f acc00 = {}, acc01 = {}, acc10 = {}, acc11 = {};
  for (int k0 = 0; k0 < K; k0 += 32) {
    v8bf a0l = *reinterpret_cast<const v8bf*>(arow0 + k0 + ka);
    v8bf a0h = *reinterpret_cast<const v8bf*>(arow0 + k0 + ka + 16);
    v8bf a1l = *reinterpret_cast<const v8bf*>(arow1 + k0 + ka);
    v8bf a1h = *reinterpret_cast<const v8bf*>(arow1 + k0 + ka + 16);
    v16bf a0 = __builtin_shufflevector(a0l, a0h, 0,1,2,3,4,5,6,7,8,9,10,11,12,13,14,15);
    v16bf a1 = __builtin_shufflevector(a1l, a1h, 0,1,2,3,4,5,6,7,8,9,10,11,12,13,14,15);
    v16bf b0 = __builtin_shufflevector(*reinterpret_cast<const v8bf*>(brow0 + k0 + ka), *reinterpret_cast<const v8bf*>(brow0 + k0 + ka + 16), 0,1,2,3,4,5,6,7,8,9,10,11,12,13,14,15);
    v16bf b1 = __builtin_shufflevector(*reinterpret_cast<const v8bf*>(brow1 + k0 + ka), *reinterpret_cast<const v8bf*>(brow1 + k0 + ka + 16), 0,1,2,3,4,5,6,7,8,9,10,11,12,13,14,15);
    acc00 = __builtin_amdgcn_wmma_f32_16x16x32_bf16(false, a0, false, b0, (short)0, acc00, false, false);
    acc01 = __builtin_amdgcn_wmma_f32_16x16x32_bf16(false, a0, false, b1, (short)0, acc01, false, false);
    acc10 = __builtin_amdgcn_wmma_f32_16x16x32_bf16(false, a1, false, b0, (short)0, acc10, false, false);
    acc11 = __builtin_amdgcn_wmma_f32_16x16x32_bf16(false, a1, false, b1, (short)0, acc11, false, false);
  }
  // C/D 16x16 f32 layout: column n = lane&15, rows m = v + (lane<16 ? 0 : 8)
  int mb = hi ? 8 : 0;
  float* c00 = C + (size_t)(tm + mb) * N + tn + r;
  float* c10 = C + (size_t)(tm + 16 + mb) * N + tn + r;
#pragma unroll
  for (int v = 0; v < 8; ++v) {
    c00[(size_t)v * N]      = acc00[v];
    c00[(size_t)v * N + 16] = acc01[v];
    c10[(size_t)v * N]      = acc10[v];
    c10[(size_t)v * N + 16] = acc11[v];
  }
}

// ---------------- GAT attention pieces ----------------
// out[i*H+h] = dot(t[i, h*F : (h+1)*F], a[h*F : ...])   (one wave per (i,h))
__global__ void k_attn_coef(const float* __restrict__ t, const float* __restrict__ a,
                            float* __restrict__ out, int Nn, int H, int F) {
  int wid = (blockIdx.x * blockDim.x + threadIdx.x) >> 5;
  int lane = threadIdx.x & 31;
  if (wid >= Nn * H) return;
  int i = wid / H, h = wid - i * H;
  const float* tp = t + (size_t)i * H * F + (size_t)h * F;
  const float* ap = a + (size_t)h * F;
  float s = 0.f;
  for (int f = lane; f < F; f += 32) s += tp[f] * ap[f];
  s = wave_sum(s);
  if (lane == 0) out[wid] = s;
}

__device__ __forceinline__ void atomic_max_f32(float* addr, float val) {
  // monotonic int-order trick; buffer initialized to -inf (0xFF800000)
  if (val >= 0.f) atomicMax((int*)addr, __float_as_int(val));
  else            atomicMin((unsigned int*)addr, __float_as_uint(val));
}

__global__ void k_edge_max(const int* __restrict__ src, const int* __restrict__ dst,
                           const float* __restrict__ el, const float* __restrict__ er,
                           float* __restrict__ ebuf, float* __restrict__ m, int H) {
  int idx = blockIdx.x * blockDim.x + threadIdx.x;
  if (idx >= NEDGE * H) return;
  int e = idx / H, h = idx - e * H;
  float v = el[src[e] * H + h] + er[dst[e] * H + h];
  v = (v > 0.f) ? v : NSLOPE * v;          // leaky_relu
  ebuf[idx] = v;
  atomic_max_f32(&m[dst[e] * H + h], v);
}

__global__ void k_edge_expsum(const int* __restrict__ dst, float* __restrict__ ebuf,
                              const float* __restrict__ m, float* __restrict__ s, int H) {
  int idx = blockIdx.x * blockDim.x + threadIdx.x;
  if (idx >= NEDGE * H) return;
  int e = idx / H, h = idx - e * H;
  float ex = __expf(ebuf[idx] - m[dst[e] * H + h]);
  ebuf[idx] = ex;
  atomicAdd(&s[dst[e] * H + h], ex);
}

// ebuf[e,h] <- ebuf[e,h] / (s[dst]+eps) / H   (alpha, head-mean folded in)
__global__ void k_edge_alpha(const int* __restrict__ dst, float* __restrict__ ebuf,
                             const float* __restrict__ s, int H) {
  int idx = blockIdx.x * blockDim.x + threadIdx.x;
  if (idx >= NEDGE * H) return;
  int e = idx / H, h = idx - e * H;
  ebuf[idx] = ebuf[idx] / (s[dst[e] * H + h] + 1e-9f) * (1.f / (float)H);
}

// out[dst, f..f+3] += sum_h alpha[e,h] * tsrc[src, h*F+f..]; float4 gathers, f32 atomics
__global__ void k_edge_aggregate(const int* __restrict__ src, const int* __restrict__ dst,
                                 const float* __restrict__ alpha,
                                 const float* __restrict__ tsrc, float* __restrict__ out,
                                 int H, int F) {
  int fq = F >> 2;
  int idx = blockIdx.x * blockDim.x + threadIdx.x;
  if (idx >= NEDGE * fq) return;
  int e = idx / fq, f4 = (idx - e * fq) << 2;
  int si = src[e], di = dst[e];
  float ax = 0.f, ay = 0.f, az = 0.f, aw = 0.f;
  for (int h = 0; h < H; ++h) {
    float a = alpha[e * H + h];
    const float4 t = *reinterpret_cast<const float4*>(
        tsrc + (size_t)si * H * F + (size_t)h * F + f4);
    ax += a * t.x; ay += a * t.y; az += a * t.z; aw += a * t.w;
  }
  float* o = out + (size_t)di * F + f4;
  atomicAdd(o + 0, ax);
  atomicAdd(o + 1, ay);
  atomicAdd(o + 2, az);
  atomicAdd(o + 3, aw);
}

// ---------------- semantic attention ----------------
// wsum[r] += (1/Nn) * tanh(proj[i,:] + bs) . q     (one wave per node)
__global__ void k_sem_score(const float* __restrict__ proj, const float* __restrict__ bs,
                            const float* __restrict__ q, float* __restrict__ wsum, int Nn) {
  int wid = (blockIdx.x * blockDim.x + threadIdx.x) >> 5;
  int lane = threadIdx.x & 31;
  if (wid >= Nn) return;
  const float* p = proj + (size_t)wid * SEMD;
  float s = 0.f;
  for (int k = lane; k < SEMD; k += 32) s += tanhf(p[k] + bs[k]) * q[k];
  s = wave_sum(s);
  if (lane == 0) atomicAdd(wsum, s / (float)Nn);
}

__global__ void k_sem_combine(const float* __restrict__ h0, const float* __restrict__ h1,
                              const float* __restrict__ wsum, float* __restrict__ outf,
                              u16* __restrict__ outb, int n, int relu) {
  int i = blockIdx.x * blockDim.x + threadIdx.x;
  if (i >= n) return;
  float w0 = wsum[0], w1 = wsum[1];
  float mx = fmaxf(w0, w1);
  float e0 = __expf(w0 - mx), e1 = __expf(w1 - mx);
  float b0 = e0 / (e0 + e1);
  float v = b0 * h0[i] + (1.f - b0) * h1[i];
  if (relu) v = fmaxf(v, 0.f);
  if (outf) outf[i] = v;
  if (outb) outb[i] = f2bf(v);
}

// ---------------- final edge scoring (one wave per edge, one float4 per lane) ----------------
__global__ void k_score(const float* __restrict__ hd, const float* __restrict__ hp,
                        const int* __restrict__ si, const int* __restrict__ di,
                        float* __restrict__ out, int n, int F) {
  int wid = (blockIdx.x * blockDim.x + threadIdx.x) >> 5;
  int lane = threadIdx.x & 31;
  if (wid >= n) return;
  const float* a = hd + (size_t)si[wid] * F;
  const float* b = hp + (size_t)di[wid] * F;
  float s = 0.f;
  for (int f = lane * 4; f < F; f += 128) {
    const float4 av = *reinterpret_cast<const float4*>(a + f);
    const float4 bv = *reinterpret_cast<const float4*>(b + f);
    s += av.x * bv.x + av.y * bv.y + av.z * bv.z + av.w * bv.w;
  }
  s = wave_sum(s);
  if (lane == 0) out[wid] = s;
}

// ---------------- host orchestration ----------------
static void launch_gemm(const u16* A, const u16* Bt, float* C, int M, int N, int K,
                        hipStream_t st) {
  long waves = (long)(M / 32) * (N / 32);   // M, N multiples of 32 for all our shapes
  long threads = waves * 32;
  k_gemm_bf16<<<(int)((threads + 255) / 256), 256, 0, st>>>(A, Bt, C, M, N, K);
}

static void run_relation(hipStream_t st,
                         const u16* xs_bf, const u16* xd_bf, int n_src, int n_dst,
                         const float* W, const float* al, const float* ar,
                         const int* src, const int* dst,
                         int K, int H, int F,
                         u16* wt, float* ts, float* td, float* el, float* er,
                         float* mb, float* sb, float* eb, float* outbuf) {
  int N = H * F;
  k_transpose_bf16<<<ceil_div(K * N, 256), 256, 0, st>>>(W, wt, K, N);
  launch_gemm(xs_bf, wt, ts, n_src, N, K, st);
  const float* tdst = ts;
  if (xd_bf != xs_bf) { launch_gemm(xd_bf, wt, td, n_dst, N, K, st); tdst = td; }
  k_attn_coef<<<ceil_div(n_src * H * 32, 256), 256, 0, st>>>(ts, al, el, n_src, H, F);
  k_attn_coef<<<ceil_div(n_dst * H * 32, 256), 256, 0, st>>>(tdst, ar, er, n_dst, H, F);
  k_fill<<<ceil_div(n_dst * H, 256), 256, 0, st>>>(mb, -INFINITY, n_dst * H);
  hipMemsetAsync(sb, 0, (size_t)n_dst * H * sizeof(float), st);
  k_edge_max<<<ceil_div(NEDGE * H, 256), 256, 0, st>>>(src, dst, el, er, eb, mb, H);
  k_edge_expsum<<<ceil_div(NEDGE * H, 256), 256, 0, st>>>(dst, eb, mb, sb, H);
  k_edge_alpha<<<ceil_div(NEDGE * H, 256), 256, 0, st>>>(dst, eb, sb, H);
  hipMemsetAsync(outbuf, 0, (size_t)n_dst * F * sizeof(float), st);
  k_edge_aggregate<<<ceil_div(NEDGE * (F / 4), 256), 256, 0, st>>>(src, dst, eb, ts, outbuf, H, F);
}

static void run_sem(hipStream_t st, float* o0, float* o1, int Nn, int F,
                    const float* Wsem, const float* bs, const float* q,
                    u16* wt, u16* obf, float* proj, float* wsum,
                    float* outf, u16* outb, int relu) {
  hipMemsetAsync(wsum, 0, 2 * sizeof(float), st);
  k_transpose_bf16<<<ceil_div(F * SEMD, 256), 256, 0, st>>>(Wsem, wt, F, SEMD);
  float* os[2] = { o0, o1 };
  for (int r = 0; r < 2; ++r) {
    k_f32_to_bf16<<<ceil_div(Nn * F, 256), 256, 0, st>>>(os[r], obf, Nn * F);
    launch_gemm(obf, wt, proj, Nn, SEMD, F, st);
    k_sem_score<<<ceil_div(Nn * 32, 256), 256, 0, st>>>(proj, bs, q, wsum + r, Nn);
  }
  k_sem_combine<<<ceil_div(Nn * F, 256), 256, 0, st>>>(o0, o1, wsum, outf, outb, Nn * F, relu);
}

extern "C" void kernel_launch(void* const* d_in, const int* in_sizes, int n_in,
                              void* d_out, int out_size, void* d_ws, size_t ws_size,
                              hipStream_t stream) {
  (void)in_sizes; (void)n_in; (void)out_size; (void)ws_size;
  const float* x_drug = (const float*)d_in[0];
  const float* x_prot = (const float*)d_in[1];
  const int* src[4] = { (const int*)d_in[2], (const int*)d_in[4],
                        (const int*)d_in[6], (const int*)d_in[8] };   // dd, dp, pd, pp
  const int* dst[4] = { (const int*)d_in[3], (const int*)d_in[5],
                        (const int*)d_in[7], (const int*)d_in[9] };
  const int* neg_src = (const int*)d_in[10];
  const int* neg_dst = (const int*)d_in[11];
  // params flattened (jax tree: dict keys sorted):
  // 12..23 l1.{dd,dp,pd,pp}.{W,al,ar} ; 24..35 l2.{...} ;
  // 36..41 sem1.{drug,protein}.{Ws,bs,q} ; 42..47 sem2.{...}
  const float *l1W[4], *l1al[4], *l1ar[4], *l2W[4], *l2al[4], *l2ar[4];
  for (int r = 0; r < 4; ++r) {
    l1W[r]  = (const float*)d_in[12 + 3 * r];
    l1al[r] = (const float*)d_in[13 + 3 * r];
    l1ar[r] = (const float*)d_in[14 + 3 * r];
    l2W[r]  = (const float*)d_in[24 + 3 * r];
    l2al[r] = (const float*)d_in[25 + 3 * r];
    l2ar[r] = (const float*)d_in[26 + 3 * r];
  }
  const float* s1Ws[2] = { (const float*)d_in[36], (const float*)d_in[39] };
  const float* s1bs[2] = { (const float*)d_in[37], (const float*)d_in[40] };
  const float* s1q [2] = { (const float*)d_in[38], (const float*)d_in[41] };
  const float* s2Ws[2] = { (const float*)d_in[42], (const float*)d_in[45] };
  const float* s2bs[2] = { (const float*)d_in[43], (const float*)d_in[46] };
  const float* s2q [2] = { (const float*)d_in[44], (const float*)d_in[47] };

  // -------- workspace carve-up (~330 MB) --------
  char* base = (char*)d_ws; size_t off = 0;
  auto alloc = [&](size_t bytes) -> void* {
    void* p = base + off; off = (off + bytes + 255) & ~(size_t)255; return p;
  };
  u16*   xd_bf  = (u16*)  alloc((size_t)NDRUG * F_IN * 2);
  u16*   xp_bf  = (u16*)  alloc((size_t)NPROT * F_IN * 2);
  u16*   wt     = (u16*)  alloc((size_t)768 * 512 * 2);          // reused transposed-B
  float* ts     = (float*)alloc((size_t)NDRUG * 768 * 4);        // src transform
  float* td     = (float*)alloc((size_t)NPROT * 768 * 4);        // dst transform
  float* el     = (float*)alloc((size_t)NDRUG * 3 * 4);
  float* er     = (float*)alloc((size_t)NPROT * 3 * 4);
  float* mb     = (float*)alloc((size_t)NDRUG * 3 * 4);
  float* sb     = (float*)alloc((size_t)NDRUG * 3 * 4);
  float* eb     = (float*)alloc((size_t)NEDGE * 3 * 4);
  float* g1[4]; for (int r = 0; r < 4; ++r) g1[r] = (float*)alloc((size_t)NDRUG * F_HID * 4);
  u16*   h1d_bf = (u16*)  alloc((size_t)NDRUG * F_HID * 2);
  u16*   h1p_bf = (u16*)  alloc((size_t)NPROT * F_HID * 2);
  float* g2[4]; for (int r = 0; r < 4; ++r) g2[r] = (float*)alloc((size_t)NDRUG * F_OUT * 4);
  float* h2d    = (float*)alloc((size_t)NDRUG * F_OUT * 4);
  float* h2p    = (float*)alloc((size_t)NPROT * F_OUT * 4);
  float* wsum   = (float*)alloc(256);
  u16*   obf    = (u16*)ts;     // reuse big transform buffers between phases
  float* proj   = td;

  // -------- quantize inputs to bf16 --------
  k_f32_to_bf16<<<ceil_div(NDRUG * F_IN, 256), 256, 0, stream>>>(x_drug, xd_bf, NDRUG * F_IN);
  k_f32_to_bf16<<<ceil_div(NPROT * F_IN, 256), 256, 0, stream>>>(x_prot, xp_bf, NPROT * F_IN);

  // -------- layer 1 (H=3, F=256), rel order dd,dp,pd,pp --------
  const u16* sA1[4] = { xd_bf, xd_bf, xp_bf, xp_bf };
  const u16* dA1[4] = { xd_bf, xp_bf, xd_bf, xp_bf };
  int nsrc[4] = { NDRUG, NDRUG, NPROT, NPROT };
  int ndst[4] = { NDRUG, NPROT, NDRUG, NPROT };
  for (int r = 0; r < 4; ++r)
    run_relation(stream, sA1[r], dA1[r], nsrc[r], ndst[r], l1W[r], l1al[r], l1ar[r],
                 src[r], dst[r], F_IN, 3, F_HID, wt, ts, td, el, er, mb, sb, eb, g1[r]);

  // semantic attention + relu + bf16 requant: drug <- [dd, pd], protein <- [dp, pp]
  run_sem(stream, g1[0], g1[2], NDRUG, F_HID, s1Ws[0], s1bs[0], s1q[0],
          wt, obf, proj, wsum, nullptr, h1d_bf, 1);
  run_sem(stream, g1[1], g1[3], NPROT, F_HID, s1Ws[1], s1bs[1], s1q[1],
          wt, obf, proj, wsum, nullptr, h1p_bf, 1);

  // -------- layer 2 (H=1, F=128) --------
  const u16* sA2[4] = { h1d_bf, h1d_bf, h1p_bf, h1p_bf };
  const u16* dA2[4] = { h1d_bf, h1p_bf, h1d_bf, h1p_bf };
  for (int r = 0; r < 4; ++r)
    run_relation(stream, sA2[r], dA2[r], nsrc[r], ndst[r], l2W[r], l2al[r], l2ar[r],
                 src[r], dst[r], F_HID, 1, F_OUT, wt, ts, td, el, er, mb, sb, eb, g2[r]);

  run_sem(stream, g2[0], g2[2], NDRUG, F_OUT, s2Ws[0], s2bs[0], s2q[0],
          wt, obf, proj, wsum, h2d, nullptr, 0);
  run_sem(stream, g2[1], g2[3], NPROT, F_OUT, s2Ws[1], s2bs[1], s2q[1],
          wt, obf, proj, wsum, h2p, nullptr, 0);

  // -------- scoring: pos uses (src_dp, dst_dp), neg uses negatives --------
  float* out = (float*)d_out;
  k_score<<<ceil_div(NEDGE * 32, 256), 256, 0, stream>>>(h2d, h2p, src[1], dst[1],
                                                         out, NEDGE, F_OUT);
  k_score<<<ceil_div(NEDGE * 32, 256), 256, 0, stream>>>(h2d, h2p, neg_src, neg_dst,
                                                         out + NEDGE, NEDGE, F_OUT);
}
